// NodeMLPActor_43997644980641
// MI455X (gfx1250) — matrix operand order repaired and checked by reading, not compiled, not run on hardware
//
#include <hip/hip_runtime.h>

#define INPUT_DIM 64
#define HIDDEN    256
#define M_TOTAL   (256 * 2048)          // BATCH * N_NODES rows
#define ROWS_PER_BLOCK 128              // 8 waves * 16 rows
#define NCHUNKS   (M_TOTAL / ROWS_PER_BLOCK)   // 4096
#define GRID      1024
#define MIN_VALF  (-1e8f)
#define BN_EPSF   (1e-5f)

typedef __attribute__((ext_vector_type(16))) _Float16 v16h;
typedef __attribute__((ext_vector_type(8)))  _Float16 v8h;
typedef __attribute__((ext_vector_type(8)))  float    v8f;
typedef __attribute__((ext_vector_type(4)))  float    v4f;

// ---- dynamic LDS layout (bytes) ----
#define OFF_W1T   0                         // [256][64]  f16 : 32768
#define OFF_W2T   (OFF_W1T + 32768)         // [256][256] f16 : 131072
#define OFF_HT    (OFF_W2T + 131072)        // 8 waves * 16*256 f16 : 65536
#define OFF_BNS   (OFF_HT + 65536)          // 256 f32
#define OFF_BNT   (OFF_BNS + 1024)
#define OFF_B2    (OFF_BNT + 1024)
#define OFF_W3    (OFF_B2 + 1024)
#define LDS_BYTES (OFF_W3 + 1024)           // 233472 B < 320 KB/WGP

__global__ void __launch_bounds__(256)
node_mlp_wmma_kernel(const float* __restrict__ obs,   const int* __restrict__ mask,
                     const float* __restrict__ W1,    const float* __restrict__ b1,
                     const float* __restrict__ gamma, const float* __restrict__ beta,
                     const float* __restrict__ rmean, const float* __restrict__ rvar,
                     const float* __restrict__ W2,    const float* __restrict__ b2,
                     const float* __restrict__ W3,    const float* __restrict__ b3,
                     float* __restrict__ out)
{
    extern __shared__ __align__(32) char smem[];
    _Float16* W1T = (_Float16*)(smem + OFF_W1T);
    _Float16* W2T = (_Float16*)(smem + OFF_W2T);
    _Float16* HT  = (_Float16*)(smem + OFF_HT);
    float* bns = (float*)(smem + OFF_BNS);
    float* bnt = (float*)(smem + OFF_BNT);
    float* b2s = (float*)(smem + OFF_B2);
    float* w3s = (float*)(smem + OFF_W3);

    const int tid     = threadIdx.x;
    const int lane    = tid & 31;
    const int wave    = tid >> 5;
    const int halfrow = lane & 15;        // A-row / B-col within 16
    const int koff    = (lane >> 4) * 8;  // unused; A and B fragments both use khi
    const int khi     = (lane >> 4) * 16; // B-fragment K sub-offset
    const int m0      = (lane >> 4) * 8;  // C rows base for this half-wave

    // ---- stage weights into LDS (f32 -> f16, transposed to K-contiguous) ----
    for (int idx = tid; idx < INPUT_DIM * HIDDEN; idx += 256) {
        int k = idx >> 8, n = idx & 255;                  // W1 is [64][256] row-major
        W1T[n * INPUT_DIM + k] = (_Float16)W1[idx];
    }
    for (int idx = tid; idx < HIDDEN * HIDDEN; idx += 256) {
        int k = idx >> 8, n = idx & 255;                  // W2 is [256][256] row-major
        W2T[n * HIDDEN + k] = (_Float16)W2[idx];
    }
    if (tid < HIDDEN) {
        int n = tid;
        float s = gamma[n] * rsqrtf(rvar[n] + BN_EPSF);
        bns[n] = s;
        bnt[n] = (b1[n] - rmean[n]) * s + beta[n];        // fold b1 + BN shift
        b2s[n] = b2[n];
        w3s[n] = W3[n];
    }
    __syncthreads();

    const float b3v = b3[0];
    _Float16* HTw = HT + wave * 16 * HIDDEN;              // wave-private staging tile

    for (int chunk = blockIdx.x; chunk < NCHUNKS; chunk += GRID) {
        const long long rowBase = (long long)chunk * ROWS_PER_BLOCK + wave * 16;
        const float* xrow = obs + (rowBase + halfrow) * INPUT_DIM;

        // speculative prefetch of next chunk's rows (global_prefetch_b8)
        if (chunk + GRID < NCHUNKS) {
            const float* nx = obs +
                ((long long)(chunk + GRID) * ROWS_PER_BLOCK + wave * 16 + halfrow) * INPUT_DIM;
            __builtin_prefetch(nx, 0, 1);
        }

        // ---- GEMM1 A fragments straight from global, f32 -> f16 ----
        v16h a0, a1;
        {
            const v4f* q = (const v4f*)(xrow + khi);        // K: khi..khi+15
            v4f p0 = q[0], p1 = q[1], p2 = q[2], p3 = q[3];
            const v4f* r = (const v4f*)(xrow + 32 + khi);   // second K-step (k0 = 32)
            v4f r0 = r[0], r1 = r[1], r2 = r[2], r3 = r[3];
            #pragma unroll
            for (int i = 0; i < 4; ++i) {
                a0[i]      = (_Float16)p0[i];  a0[i + 4]  = (_Float16)p1[i];
                a0[i + 8]  = (_Float16)p2[i];  a0[i + 12] = (_Float16)p3[i];
                a1[i]      = (_Float16)r0[i];  a1[i + 4]  = (_Float16)r1[i];
                a1[i + 8]  = (_Float16)r2[i];  a1[i + 12] = (_Float16)r3[i];
            }
        }

        // ---- GEMM1 (16x64 @ 64x256) + BN affine + ReLU -> HT(f16) ----
        #pragma unroll
        for (int j = 0; j < 16; ++j) {
            const int n = j * 16 + halfrow;
            const _Float16* wrow = W1T + n * INPUT_DIM + khi;
            v16h bfrag0 = *(const v16h*)(wrow);
            v16h bfrag1 = *(const v16h*)(wrow + 32);
            v8f c = {};
            c = __builtin_amdgcn_wmma_f32_16x16x32_f16(false, a0, false, bfrag0, (short)0, c, false, false);
            c = __builtin_amdgcn_wmma_f32_16x16x32_f16(false, a1, false, bfrag1, (short)0, c, false, false);
            const float s = bns[n], t = bnt[n];
            #pragma unroll
            for (int r8 = 0; r8 < 8; ++r8) {
                float h = fmaxf(c[r8] * s + t, 0.0f);
                HTw[(m0 + r8) * HIDDEN + n] = (_Float16)h;   // wave-local, DS in-order
            }
        }

        // ---- GEMM2 A fragments from LDS (cached across all 16 N-tiles) ----
        v16h a2[8];
        #pragma unroll
        for (int k = 0; k < 8; ++k) {
            const _Float16* hrow = HTw + halfrow * HIDDEN + k * 32 + khi;
            *(v8h*)&a2[k]       = *(const v8h*)(hrow);
            ((v8h*)&a2[k])[1]   = *(const v8h*)(hrow + 8);
        }

        // ---- GEMM2 (16x256 @ 256x256) + bias + ReLU, fused GEMM3 dot ----
        v8f yacc = {};
        #pragma unroll
        for (int j = 0; j < 16; ++j) {
            const int n = j * 16 + halfrow;
            const float bias = b2s[n];
            v8f c = {bias, bias, bias, bias, bias, bias, bias, bias};
            const _Float16* wrow = W2T + n * HIDDEN + khi;
            #pragma unroll
            for (int k = 0; k < 8; ++k) {
                v16h bfrag = *(const v16h*)(wrow + k * 32);
                c = __builtin_amdgcn_wmma_f32_16x16x32_f16(false, a2[k], false, bfrag, (short)0, c, false, false);
            }
            const float w3n = w3s[n];
            #pragma unroll
            for (int r8 = 0; r8 < 8; ++r8)
                yacc[r8] += fmaxf(c[r8], 0.0f) * w3n;        // GEMM3 partial (per-lane column)
        }

        // ---- reduce over the 16 columns held by each half-wave ----
        #pragma unroll
        for (int off = 8; off >= 1; off >>= 1) {
            #pragma unroll
            for (int r8 = 0; r8 < 8; ++r8)
                yacc[r8] += __shfl_xor(yacc[r8], off, 32);
        }

        if (halfrow == 0) {                                   // lanes 0 and 16 write 8 rows each
            #pragma unroll
            for (int r8 = 0; r8 < 8; ++r8) {
                long long row = rowBase + m0 + r8;
                float y = yacc[r8] + b3v;
                out[row] = (mask[row] != 0) ? y : MIN_VALF;   // scatter-with-fill
            }
        }
    }
}

extern "C" void kernel_launch(void* const* d_in, const int* in_sizes, int n_in,
                              void* d_out, int out_size, void* d_ws, size_t ws_size,
                              hipStream_t stream) {
    (void)in_sizes; (void)n_in; (void)d_ws; (void)ws_size; (void)out_size;
    const float* obs   = (const float*)d_in[0];
    const int*   mask  = (const int*)  d_in[1];
    const float* W1    = (const float*)d_in[2];
    const float* b1    = (const float*)d_in[3];
    const float* gamma = (const float*)d_in[4];
    const float* beta  = (const float*)d_in[5];
    const float* rmean = (const float*)d_in[6];
    const float* rvar  = (const float*)d_in[7];
    const float* W2    = (const float*)d_in[8];
    const float* b2    = (const float*)d_in[9];
    const float* W3    = (const float*)d_in[10];
    const float* b3    = (const float*)d_in[11];
    float* out = (float*)d_out;

    node_mlp_wmma_kernel<<<GRID, 256, LDS_BYTES, stream>>>(
        obs, mask, W1, b1, gamma, beta, rmean, rvar, W2, b2, W3, b3, out);
}
